// MMDLayer_51917564674624
// MI455X (gfx1250) — matrix operand order repaired and checked by reading, not compiled, not run on hardware
//
#include <hip/hip_runtime.h>

typedef __attribute__((ext_vector_type(16))) _Float16 v16h;
typedef __attribute__((ext_vector_type(8)))  float    v8f;

#define TILE_M 128
#define TILE_N 128
#define KDIM   64
#define LDS_K  72   // padded LDS row stride in halves (144 B) -> conflict-free b128 frag loads

union Frag { v16h h; uint4 u[2]; };

__device__ __forceinline__ float wave_reduce(float s) {
#pragma unroll
  for (int off = 16; off > 0; off >>= 1)
    s += __shfl_xor(s, off, 32);
  return s;
}

// ---------------------------------------------------------------------------
// One-time f32 -> f16 conversion of z into workspace (each element of z is
// reused by ~256 tile blocks; convert once instead of per-tile).
__global__ __launch_bounds__(256)
void convert_f16_kernel(const float* __restrict__ X, _Float16* __restrict__ Xh) {
  int idx = blockIdx.x * blockDim.x + threadIdx.x;   // float4 index
  float4 v = *(const float4*)(X + (size_t)idx * 4);
  union { _Float16 h[4]; uint2 u; } p;
  p.h[0] = (_Float16)v.x; p.h[1] = (_Float16)v.y;
  p.h[2] = (_Float16)v.z; p.h[3] = (_Float16)v.w;
  *(uint2*)(Xh + (size_t)idx * 4) = p.u;
}

// Row squared-norms in exact fp32, pre-scaled by log2e/4096 so the epilogue
// is a single FMA per pair.
__global__ __launch_bounds__(256)
void row_norms_kernel(const float* __restrict__ X, float* __restrict__ Nrm,
                      int nrows, float nscale) {
  int i = blockIdx.x * blockDim.x + threadIdx.x;
  if (i >= nrows) return;
  const float* row = X + (size_t)i * KDIM;
  float s = 0.f;
#pragma unroll
  for (int k = 0; k < KDIM; ++k) { float v = row[k]; s = fmaf(v, v, s); }
  Nrm[i] = s * nscale;
}

// Pad gaussian 200 -> 256 rows as f16: padded rows are zero data + huge
// scaled norm, so exp2(-huge) == 0 and they contribute nothing to any sum.
__global__ __launch_bounds__(256)
void pad_gauss_kernel(const float* __restrict__ G, _Float16* __restrict__ Gh,
                      float* __restrict__ Gn, float nscale) {
  int i = threadIdx.x; // 256 threads, 1 row each
  if (i < 200) {
    float s = 0.f;
#pragma unroll
    for (int k = 0; k < KDIM; ++k) {
      float v = G[i * KDIM + k];
      Gh[i * KDIM + k] = (_Float16)v;
      s = fmaf(v, v, s);
    }
    Gn[i] = s * nscale;
  } else {
#pragma unroll
    for (int k = 0; k < KDIM; ++k) Gh[i * KDIM + k] = (_Float16)0.f;
    Gn[i] = 1.0e5f; // exp2(-1e5) underflows to exactly 0
  }
}

// ---------------------------------------------------------------------------
// Tile kernel: sum_{i,j in 128-blocks} exp2(2s*A_i.B_j - ni' - nj')
// A,B: f16 row-major (stride 64). Tiles staged with CDNA5 async global->LDS
// DMA (ASYNCcnt), fragments fed to v_wmma_f32_16x16x32_f16.
__global__ __launch_bounds__(256)
void mmd_tile_kernel(const _Float16* __restrict__ A, const float* __restrict__ An,
                     const _Float16* __restrict__ B, const float* __restrict__ Bn,
                     float* __restrict__ partial, float s2) {
  __shared__ _Float16 As[TILE_M][LDS_K];
  __shared__ _Float16 Bs[TILE_N][LDS_K];
  __shared__ float AnS[TILE_M];
  __shared__ float BnS[TILE_N];
  __shared__ float red[8];

  const int t  = threadIdx.x;
  const int ib = blockIdx.y * TILE_M;
  const int jb = blockIdx.x * TILE_N;

  // Async-stage both 128x64 f16 tiles: 1024 16-byte chunks each, 4 per thread
  // per tile, written straight into LDS by the memory pipe (no VGPR bounce).
  const uint64_t abase = (uint64_t)(uintptr_t)(A + (size_t)ib * KDIM);
  const uint64_t bbase = (uint64_t)(uintptr_t)(B + (size_t)jb * KDIM);
#pragma unroll
  for (int q = 0; q < 4; ++q) {
    int idx = t + q * 256;       // 8-half chunk index
    int r   = idx >> 3;          // row in tile
    int c8  = (idx & 7) << 3;    // starting half within row
    uint32_t la = (uint32_t)(uintptr_t)&As[r][c8];  // flat low-32 == LDS offset
    uint64_t ga = abase + (uint64_t)(r * KDIM + c8) * 2;
    asm volatile("global_load_async_to_lds_b128 %0, %1, off"
                 :: "v"(la), "v"(ga) : "memory");
    uint32_t lb = (uint32_t)(uintptr_t)&Bs[r][c8];
    uint64_t gb = bbase + (uint64_t)(r * KDIM + c8) * 2;
    asm volatile("global_load_async_to_lds_b128 %0, %1, off"
                 :: "v"(lb), "v"(gb) : "memory");
  }
  if (t < 128) AnS[t] = An[ib + t];
  else         BnS[t - 128] = Bn[jb + (t - 128)];
  asm volatile("s_wait_asynccnt 0x0" ::: "memory");
  __syncthreads();

  const int wave  = t >> 5;
  const int lane  = t & 31;
  const int wm    = wave >> 1;        // 0..3  -> 32-row slab
  const int wn    = wave & 1;         // 0..1  -> 64-col slab
  const int mbase = wm * 32;
  const int nbase = wn * 64;
  const int lr    = lane & 15;
  const int hi    = lane >> 4;        // lane half selects K-group

  v8f c[2][4] = {};

#pragma unroll
  for (int kc = 0; kc < KDIM; kc += 32) {
    Frag a[2];
#pragma unroll
    for (int mi = 0; mi < 2; ++mi) {
      int r  = mbase + mi * 16 + lr;
      int k0 = kc + hi * 8;
      // ISA 16-bit A 16x32 layout: lanes 0-15 K{0-7,16-23}, lanes 16-31 K{8-15,24-31}
      a[mi].u[0] = *(const uint4*)&As[r][k0];
      a[mi].u[1] = *(const uint4*)&As[r][k0 + 16];
    }
    Frag b[4];
#pragma unroll
    for (int ni = 0; ni < 4; ++ni) {
      int n  = nbase + ni * 16 + lr;
      int kb = kc + (hi << 3);
      // ISA 16-bit B 32x16 layout: lanes 0-15 K{0-7,16-23}, lanes 16-31 K{8-15,24-31}
      b[ni].u[0] = *(const uint4*)&Bs[n][kb];
      b[ni].u[1] = *(const uint4*)&Bs[n][kb + 16];
    }
#pragma unroll
    for (int mi = 0; mi < 2; ++mi)
#pragma unroll
      for (int ni = 0; ni < 4; ++ni)
        c[mi][ni] = __builtin_amdgcn_wmma_f32_16x16x32_f16(
            false, a[mi].h, false, b[ni].h, (short)0, c[mi][ni], false, false);
  }

  // Epilogue: arg = fma(dot, 2*scale, -(ni'+nj')); one FMA + exp2 + add / pair.
  // C layout: col = lane&15, rows = (lane>>4)*8 + v.
  float sum = 0.f;
#pragma unroll
  for (int mi = 0; mi < 2; ++mi) {
    int rbase = mbase + mi * 16 + hi * 8;
    float niv[8];
#pragma unroll
    for (int v = 0; v < 8; ++v) niv[v] = AnS[rbase + v];
#pragma unroll
    for (int nt = 0; nt < 4; ++nt) {
      float nj = BnS[nbase + nt * 16 + lr];
#pragma unroll
      for (int v = 0; v < 8; ++v) {
        float arg = fmaf(c[mi][nt][v], s2, -(niv[v] + nj));
        sum += __builtin_amdgcn_exp2f(arg);   // native v_exp_f32 (TRANS, co-issues)
      }
    }
  }

  sum = wave_reduce(sum);
  if (lane == 0) red[wave] = sum;
  __syncthreads();
  if (t == 0) {
    float s = 0.f;
#pragma unroll
    for (int w = 0; w < 8; ++w) s += red[w];
    partial[blockIdx.y * gridDim.x + blockIdx.x] = s;
  }
}

// ---------------------------------------------------------------------------
// Deterministic final combine: fixed per-thread strided order + LDS tree.
__global__ __launch_bounds__(256)
void final_reduce_kernel(const float* __restrict__ pzz, int nzz,
                         const float* __restrict__ pgz, int ngz,
                         const float* __restrict__ pgg, int ngg,
                         float* __restrict__ out) {
  __shared__ float red[256];
  int t = threadIdx.x;
  const float wzz =  1.0f / (16384.0f * 16384.0f);
  const float wgz = -2.0f / (200.0f * 16384.0f);
  const float wgg =  1.0f / (200.0f * 200.0f);
  float s = 0.f;
  for (int i = t; i < nzz; i += 256) s += wzz * pzz[i];
  for (int i = t; i < ngz; i += 256) s += wgz * pgz[i];
  for (int i = t; i < ngg; i += 256) s += wgg * pgg[i];
  red[t] = s;
  __syncthreads();
  for (int off = 128; off > 0; off >>= 1) {
    if (t < off) red[t] += red[t + off];
    __syncthreads();
  }
  if (t == 0) out[0] = 10.0f * red[0];  // (1 - ALPHA - LAMBDA) = 10
}

// ---------------------------------------------------------------------------
extern "C" void kernel_launch(void* const* d_in, const int* in_sizes, int n_in,
                              void* d_out, int out_size, void* d_ws, size_t ws_size,
                              hipStream_t stream) {
  const float* z = (const float*)d_in[0];      // 16384 x 64
  const float* g = (const float*)d_in[1];      // 200 x 64
  float* out = (float*)d_out;

  // Workspace layout (f16 arrays first for 16B alignment of tile rows)
  _Float16* z_h = (_Float16*)d_ws;             // 16384*64 halves = 2 MB
  _Float16* g_h = z_h + (size_t)16384 * 64;    // 256*64 halves   = 32 KB
  float* zn  = (float*)(g_h + 256 * 64);       // 16384
  float* gn  = zn + 16384;                     // 256 (padded)
  float* pzz = gn + 256;                       // 128*128 = 16384 partials
  float* pgz = pzz + 16384;                    // 128*2 = 256 partials
  float* pgg = pgz + 256;                      // 2*2 = 4 partials

  // exp(-sq/d^2) = exp2(-sq * log2(e)/4096)
  const float nscale = 1.4426950408889634f / 4096.0f; // applied to norms
  const float s2     = 2.0f * nscale;                  // applied to dot

  convert_f16_kernel<<<1024, 256, 0, stream>>>(z, z_h);          // 16384*64/1024
  row_norms_kernel<<<64, 256, 0, stream>>>(z, zn, 16384, nscale);
  pad_gauss_kernel<<<1, 256, 0, stream>>>(g, g_h, gn, nscale);

  mmd_tile_kernel<<<dim3(128, 128), 256, 0, stream>>>(z_h, zn, z_h, zn, pzz, s2);
  mmd_tile_kernel<<<dim3(128,   2), 256, 0, stream>>>(g_h, gn, z_h, zn, pgz, s2);
  mmd_tile_kernel<<<dim3(  2,   2), 256, 0, stream>>>(g_h, gn, g_h, gn, pgg, s2);

  final_reduce_kernel<<<1, 256, 0, stream>>>(pzz, 16384, pgz, 256, pgg, 4, out);
}
